// GCNLinkPredictor_62371515072904
// MI455X (gfx1250) — matrix operand order repaired and checked by reading, not compiled, not run on hardware
//
#include <hip/hip_runtime.h>
#include <hip/hip_bf16.h>

typedef __attribute__((ext_vector_type(16))) _Float16 v16h;
typedef __attribute__((ext_vector_type(8)))  _Float16 v8h;
typedef __attribute__((ext_vector_type(8)))  float    v8f;

#define GCN_N   100000
#define GCN_E   1600000
#define F_IN    256
#define HDIM    64
#define NPOSG   (GCN_E / 16)        // 100000 groups of 16 edges per pass
#define NGROUPS (2 * NPOSG)         // pos + neg

// Wave-local LDS fence: HW LDS pipe is in-order per wave; this stops the
// compiler reordering across the type-punned LDS reuse and drains DScnt.
#define LDS_FENCE() asm volatile("s_wait_dscnt 0" ::: "memory")

static __device__ __forceinline__ v8f wmma_f16(v16h a, v16h b, v8f c) {
  return __builtin_amdgcn_wmma_f32_16x16x32_f16(false, a, false, b, (short)0, c,
                                                false, false);
}

// A fragment (16x32 f16, per ISA 7.12.2): lane holds row m=lane&15,
// chunks [kb..kb+7] and [16+kb..16+kb+7], kb = 8*(lane>=16).
static __device__ __forceinline__ v16h pack_a_from_f32(const float* __restrict__ row,
                                                       int ktile, int kb) {
  const float* p = row + ktile * 32;
  const float4 q0 = *(const float4*)(p + kb);
  const float4 q1 = *(const float4*)(p + kb + 4);
  const float4 q2 = *(const float4*)(p + 16 + kb);
  const float4 q3 = *(const float4*)(p + 16 + kb + 4);
  v16h a;
  a[0]  = (_Float16)q0.x; a[1]  = (_Float16)q0.y; a[2]  = (_Float16)q0.z; a[3]  = (_Float16)q0.w;
  a[4]  = (_Float16)q1.x; a[5]  = (_Float16)q1.y; a[6]  = (_Float16)q1.z; a[7]  = (_Float16)q1.w;
  a[8]  = (_Float16)q2.x; a[9]  = (_Float16)q2.y; a[10] = (_Float16)q2.z; a[11] = (_Float16)q2.w;
  a[12] = (_Float16)q3.x; a[13] = (_Float16)q3.y; a[14] = (_Float16)q3.z; a[15] = (_Float16)q3.w;
  return a;
}

static __device__ __forceinline__ v16h pack_a_from_lds(const _Float16* row, int ktile, int kb) {
  const _Float16* p = row + ktile * 32;
  v8h lo = *(const v8h*)(p + kb);
  v8h hi = *(const v8h*)(p + 16 + kb);
  v16h a;
#pragma unroll
  for (int i = 0; i < 8; ++i) { a[i] = lo[i]; a[8 + i] = hi[i]; }
  return a;
}

// Pre-packed B fragment: [(kt*ntiles+nt)][lane][16] halves, one 32B load/lane.
static __device__ __forceinline__ v16h load_b_frag(const _Float16* __restrict__ pk,
                                                   int ntiles, int kt, int nt, int lane) {
  return *(const v16h*)(pk + (((size_t)(kt * ntiles + nt) * 32 + lane) << 4));
}

// -------- weight packing into WMMA B layout (K rows x N cols, fp32 -> f16) ----
__global__ void pack_b_kernel(const float* __restrict__ W, int ldw, int ntiles,
                              _Float16* __restrict__ out) {
  int tid  = blockIdx.x * 256 + threadIdx.x;
  int i    = tid & 15;
  int lane = (tid >> 4) & 31;
  int rest = tid >> 9;               // kt*ntiles + nt
  int nt   = rest % ntiles;
  int kt   = rest / ntiles;
  int n    = nt * 16 + (lane & 15);
  int k    = kt * 32 + ((lane >> 4) << 3) + (i & 7) + ((i >> 3) << 4);
  out[tid] = (_Float16)W[(size_t)k * ldw + n];
}

// ------------------------------ degree kernels -------------------------------
__global__ void init_deg_kernel(float* __restrict__ deg) {
  int i = blockIdx.x * 256 + threadIdx.x;
  if (i < GCN_N) deg[i] = 1.0f;      // self loop
}
__global__ void count_deg_kernel(const int* __restrict__ ei, float* __restrict__ deg) {
  int e = blockIdx.x * 256 + threadIdx.x;          // exact grid
  atomicAdd(&deg[ei[GCN_E + e]], 1.0f);
}
__global__ void dinv_kernel(float* __restrict__ deg) {
  int i = blockIdx.x * 256 + threadIdx.x;
  if (i < GCN_N) deg[i] = rsqrtf(deg[i]);          // deg >= 1 always
}

// ------------------------------- WMMA GEMM -----------------------------------
// C[M,64] = A[M,K] (fp32, cast to f16) @ Bpack. 8 waves/block: 2 mtiles x 4 ntiles.
__global__ void gemm_wmma_kernel(const float* __restrict__ A, int lda, int ktiles,
                                 const _Float16* __restrict__ Bp,
                                 float* __restrict__ C) {
  int tid   = threadIdx.x;
  int lane  = tid & 31;
  int wid   = tid >> 5;
  int ntile = wid & 3;
  int mtile = blockIdx.x * 2 + (wid >> 2);
  int kb    = (lane >> 4) << 3;
  const float* rowPtr = A + (size_t)(mtile * 16 + (lane & 15)) * lda;
  v8f acc = {};
  for (int kt = 0; kt < ktiles; ++kt) {
    v16h a = pack_a_from_f32(rowPtr, kt, kb);
    v16h b = load_b_frag(Bp, 4, kt, ntile, lane);
    acc = wmma_f16(a, b, acc);
  }
  int col = ntile * 16 + (lane & 15);
  int row = mtile * 16 + ((lane >> 4) << 3);
#pragma unroll
  for (int r = 0; r < 8; ++r)
    C[(size_t)(row + r) * HDIM + col] = acc[r];
}

// --------------------------- GCN aggregation ---------------------------------
__global__ void selfloop_init_kernel(const float* __restrict__ t,
                                     const float* __restrict__ dinv,
                                     float* __restrict__ agg) {
  int tid = blockIdx.x * 256 + threadIdx.x;        // exact: N*64/256
  int i = tid >> 6, f = tid & 63;
  float d = dinv[i];
  agg[tid] = d * d * t[(size_t)i * HDIM + f];
}
__global__ void scatter_kernel(const float* __restrict__ t,
                               const int* __restrict__ ei,
                               const float* __restrict__ dinv,
                               float* __restrict__ agg) {
  int tid = blockIdx.x * 256 + threadIdx.x;        // exact: E*64/256
  int e = tid >> 6, f = tid & 63;
  int s = ei[e], d = ei[GCN_E + e];
  float w = dinv[s] * dinv[d];
  atomicAdd(&agg[(size_t)d * HDIM + f], w * t[(size_t)s * HDIM + f]);
}
__global__ void bias_relu_kernel(const float* __restrict__ agg,
                                 const float* __restrict__ b,
                                 float* __restrict__ z) {
  int tid = blockIdx.x * 256 + threadIdx.x;        // exact
  z[tid] = fmaxf(agg[tid] + b[tid & 63], 0.0f);
}

// ------------------------------- edge MLP ------------------------------------
// One wave per 16-edge group. Layer1 16x128 (32 WMMA), layer2 16x64 (16 WMMA),
// layer3 dot-64. LDS slice (4KB/wave) reused f16[16x128] -> f32[16x64].
__global__ void edge_mlp_kernel(const float* __restrict__ z,
                                const int* __restrict__ ei,
                                const int* __restrict__ nei,
                                const _Float16* __restrict__ pWp1a,
                                const _Float16* __restrict__ pWp1b,
                                const _Float16* __restrict__ pWp2,
                                const float* __restrict__ bp1,
                                const float* __restrict__ bp2,
                                const float* __restrict__ Wp3,
                                const float* __restrict__ bp3,
                                float* __restrict__ out) {
  __shared__ _Float16 smem[8][16 * 128];           // 32KB/block
  int tid  = threadIdx.x;
  int lane = tid & 31;
  int wid  = tid >> 5;
  int g    = blockIdx.x * 8 + wid;                 // exact grid: NGROUPS/8
  int pos  = (g < NPOSG);
  const int* e = pos ? ei : nei;
  int gb   = (pos ? g : g - NPOSG) * 16;
  int m    = lane & 15;
  int kb   = (lane >> 4) << 3;
  int rb   = (lane >> 4) << 3;                      // C-layout row base
  int srow = e[gb + m];
  int drow = e[GCN_E + gb + m];
  const float* sp = z + (size_t)srow * HDIM;
  const float* dp = z + (size_t)drow * HDIM;
  v16h As0 = pack_a_from_f32(sp, 0, kb), As1 = pack_a_from_f32(sp, 1, kb);
  v16h Ad0 = pack_a_from_f32(dp, 0, kb), Ad1 = pack_a_from_f32(dp, 1, kb);

  _Float16* h1 = smem[wid];
#pragma unroll
  for (int nt = 0; nt < 8; ++nt) {
    float bv = bp1[nt * 16 + m];
    v8f c;
#pragma unroll
    for (int r = 0; r < 8; ++r) c[r] = bv;
    c = wmma_f16(As0, load_b_frag(pWp1a, 8, 0, nt, lane), c);
    c = wmma_f16(As1, load_b_frag(pWp1a, 8, 1, nt, lane), c);
    c = wmma_f16(Ad0, load_b_frag(pWp1b, 8, 0, nt, lane), c);
    c = wmma_f16(Ad1, load_b_frag(pWp1b, 8, 1, nt, lane), c);
#pragma unroll
    for (int r = 0; r < 8; ++r)
      h1[(rb + r) * 128 + nt * 16 + m] = (_Float16)fmaxf(c[r], 0.0f);
  }
  LDS_FENCE();

  v16h Ah[4];
#pragma unroll
  for (int kt = 0; kt < 4; ++kt) Ah[kt] = pack_a_from_lds(h1 + m * 128, kt, kb);
  LDS_FENCE();

  float* h2 = (float*)smem[wid];                    // reuse slice as f32[16*64]
#pragma unroll
  for (int nt = 0; nt < 4; ++nt) {
    float bv = bp2[nt * 16 + m];
    v8f c;
#pragma unroll
    for (int r = 0; r < 8; ++r) c[r] = bv;
#pragma unroll
    for (int kt = 0; kt < 4; ++kt)
      c = wmma_f16(Ah[kt], load_b_frag(pWp2, 4, kt, nt, lane), c);
#pragma unroll
    for (int r = 0; r < 8; ++r)
      h2[(rb + r) * 64 + nt * 16 + m] = fmaxf(c[r], 0.0f);
  }
  LDS_FENCE();

  if (lane < 16) {
    const float* hr = h2 + lane * 64;
    float s = bp3[0];
#pragma unroll
    for (int k = 0; k < 64; ++k) s += hr[k] * Wp3[k];
    out[(pos ? 0 : GCN_E) + gb + lane] = s;
  }
}

// ------------------------------- launcher ------------------------------------
extern "C" void kernel_launch(void* const* d_in, const int* in_sizes, int n_in,
                              void* d_out, int out_size, void* d_ws, size_t ws_size,
                              hipStream_t stream) {
  const float* x   = (const float*)d_in[0];
  const int*   ei  = (const int*)d_in[1];
  const int*   nei = (const int*)d_in[2];
  const float* W1  = (const float*)d_in[3];
  const float* b1  = (const float*)d_in[4];
  const float* W2  = (const float*)d_in[5];
  const float* b2  = (const float*)d_in[6];
  const float* Wp1 = (const float*)d_in[7];
  const float* bp1 = (const float*)d_in[8];
  const float* Wp2 = (const float*)d_in[9];
  const float* bp2 = (const float*)d_in[10];
  const float* Wp3 = (const float*)d_in[11];
  const float* bp3 = (const float*)d_in[12];

  float* outF = (float*)d_out;
  float* posO = outF;                    // E
  float* negO = outF + GCN_E;            // E
  float* zOut = outF + 2 * GCN_E;        // N*64

  char* w = (char*)d_ws;
  size_t o = 0;
  float* t    = (float*)(w + o); o += (size_t)GCN_N * HDIM * 4;
  float* agg  = (float*)(w + o); o += (size_t)GCN_N * HDIM * 4;
  float* z1   = (float*)(w + o); o += (size_t)GCN_N * HDIM * 4;
  float* dinv = (float*)(w + o); o += (size_t)GCN_N * 4;
  _Float16* pW1   = (_Float16*)(w + o); o += (size_t)F_IN * HDIM * 2;
  _Float16* pW2   = (_Float16*)(w + o); o += (size_t)HDIM * HDIM * 2;
  _Float16* pWp1a = (_Float16*)(w + o); o += (size_t)64 * 128 * 2;
  _Float16* pWp1b = (_Float16*)(w + o); o += (size_t)64 * 128 * 2;
  _Float16* pWp2  = (_Float16*)(w + o); o += (size_t)128 * 64 * 2;

  // 1) pack weights to WMMA B-fragment layout (f16)
  pack_b_kernel<<<(F_IN * HDIM) / 256, 256, 0, stream>>>(W1, HDIM, 4, pW1);
  pack_b_kernel<<<(HDIM * HDIM) / 256, 256, 0, stream>>>(W2, HDIM, 4, pW2);
  pack_b_kernel<<<(64 * 128) / 256, 256, 0, stream>>>(Wp1, 128, 8, pWp1a);
  pack_b_kernel<<<(64 * 128) / 256, 256, 0, stream>>>(Wp1 + 64 * 128, 128, 8, pWp1b);
  pack_b_kernel<<<(128 * 64) / 256, 256, 0, stream>>>(Wp2, 64, 4, pWp2);

  // 2) degrees -> dinv
  init_deg_kernel<<<(GCN_N + 255) / 256, 256, 0, stream>>>(dinv);
  count_deg_kernel<<<GCN_E / 256, 256, 0, stream>>>(ei, dinv);
  dinv_kernel<<<(GCN_N + 255) / 256, 256, 0, stream>>>(dinv);

  // 3) layer 1: t = x@W1 ; aggregate ; z1 = relu(agg + b1)
  gemm_wmma_kernel<<<GCN_N / 32, 256, 0, stream>>>(x, F_IN, F_IN / 32, pW1, t);
  selfloop_init_kernel<<<(GCN_N * HDIM) / 256, 256, 0, stream>>>(t, dinv, agg);
  scatter_kernel<<<(GCN_E / 256) * HDIM, 256, 0, stream>>>(t, ei, dinv, agg);
  bias_relu_kernel<<<(GCN_N * HDIM) / 256, 256, 0, stream>>>(agg, b1, z1);

  // 4) layer 2: t = z1@W2 ; aggregate ; z = relu(agg + b2) -> d_out z section
  gemm_wmma_kernel<<<GCN_N / 32, 256, 0, stream>>>(z1, HDIM, HDIM / 32, pW2, t);
  selfloop_init_kernel<<<(GCN_N * HDIM) / 256, 256, 0, stream>>>(t, dinv, agg);
  scatter_kernel<<<(GCN_E / 256) * HDIM, 256, 0, stream>>>(t, ei, dinv, agg);
  bias_relu_kernel<<<(GCN_N * HDIM) / 256, 256, 0, stream>>>(agg, b2, zOut);

  // 5) edge MLP for pos + neg edges
  edge_mlp_kernel<<<NGROUPS / 8, 256, 0, stream>>>(zOut, ei, nei, pWp1a, pWp1b,
                                                   pWp2, bp1, bp2, Wp3, bp3, posO);
}
